// MemoryComponent_47820165873920
// MI455X (gfx1250) — matrix operand order repaired and checked by reading, not compiled, not run on hardware
//
#include <hip/hip_runtime.h>

#define USE_ASYNC 1
#define USE_TDM   1

#if USE_TDM && __has_builtin(__builtin_amdgcn_tensor_load_to_lds)
#define HAVE_TDM 1
#else
#define HAVE_TDM 0
#endif

// ---------------------------------------------------------------------------
// Problem constants (from reference)
// ---------------------------------------------------------------------------
#define B_     4
#define S_     2048
#define H_     1024
#define NH_    16
#define HD_    64
#define P_     64
#define INTER_ 2048
#define NROWS  (B_*S_)          // 8192
#define LR_    0.01f
#define MOM_   0.9f
#define WD_    0.01f
#define EPS_   1e-12f

typedef __bf16 bf16;
typedef __attribute__((ext_vector_type(16))) __bf16 v16bf;
typedef __attribute__((ext_vector_type(8)))  __bf16 bf16x8;
typedef __attribute__((ext_vector_type(8)))  float  v8f;
typedef __attribute__((ext_vector_type(4)))  unsigned int u32x4;
typedef __attribute__((ext_vector_type(8)))  int i32x8;
typedef __attribute__((ext_vector_type(4)))  int i32x4;

// A-matrix (16x32 bf16) per-lane K index for element e, lane-half h
__device__ __forceinline__ int akmap(int e, int h) {
    return (e < 8) ? (8*h + e) : (8 + 8*h + e);
}

// async 16B global -> LDS copy (per-lane), tracked by ASYNCcnt
__device__ __forceinline__ void async_b128(void* lds_dst, const void* gsrc) {
    unsigned l = (unsigned)(uintptr_t)lds_dst;                 // low 32b = LDS byte addr
    unsigned long long g = (unsigned long long)(uintptr_t)gsrc;
    asm volatile("global_load_async_to_lds_b128 %0, %1, off" :: "v"(l), "v"(g) : "memory");
}
__device__ __forceinline__ void async_wait0() {
    asm volatile("s_wait_asynccnt 0x0" ::: "memory");
}

// ---------------------------------------------------------------------------
// TDM: DMA a [32 x 64] bf16 tile (row stride N elems) from global into LDS,
// with D# LDS padding of 4 DWORDs per 32 DWORDs -> LDS row stride 72 elems.
// Issue from one wave; tracked by TENSORcnt.
// ---------------------------------------------------------------------------
__device__ __forceinline__ void tdm_load_tile_b(const bf16* gtile, unsigned lds_addr,
                                                unsigned n_cols, unsigned k_rows) {
#if HAVE_TDM
    unsigned long long ga = (unsigned long long)(uintptr_t)gtile;
    u32x4 g0;
    g0[0] = 1u;                                            // count=1 (valid), user mode
    g0[1] = lds_addr;                                      // LDS byte address
    g0[2] = (unsigned)(ga & 0xffffffffu);                  // global_addr[31:0]
    g0[3] = (unsigned)((ga >> 32) & 0x1ffffffu) | (2u << 30); // addr[56:32] | type=2
    i32x8 g1;
    // data_size=2B (code 1) | pad_enable | pad_interval=32dw (code 4) | pad_amount=4dw (code 3)
    g1[0] = (int)((1u << 16) | (1u << 20) | (4u << 22) | (3u << 25));
    g1[1] = (int)((n_cols & 0xffffu) << 16);               // tensor_dim0[15:0]
    g1[2] = (int)(((n_cols >> 16) & 0xffffu) | ((k_rows & 0xffffu) << 16)); // dim0 hi | dim1 lo
    g1[3] = (int)(((k_rows >> 16) & 0xffffu) | (64u << 16));                // dim1 hi | tile_dim0=64
    g1[4] = 32;                                            // tile_dim1=32, tile_dim2=0
    g1[5] = (int)n_cols;                                   // tensor_dim0_stride[31:0]
    g1[6] = 0;                                             // stride0 hi | stride1 lo
    g1[7] = 0;
    i32x4 z4 = {0, 0, 0, 0};
    i32x8 z8 = {0, 0, 0, 0, 0, 0, 0, 0};
    __builtin_amdgcn_tensor_load_to_lds(g0, g1, z4, z4, z8, 0);
#else
    (void)gtile; (void)lds_addr; (void)n_cols; (void)k_rows;
#endif
}

// ---------------------------------------------------------------------------
// f32 -> bf16 conversion
// ---------------------------------------------------------------------------
__global__ void f32_to_bf16_k(const float* __restrict__ x, bf16* __restrict__ y, int n) {
    int i = blockIdx.x * blockDim.x + threadIdx.x;
    if (i < n) y[i] = (bf16)x[i];
}

// ---------------------------------------------------------------------------
// LayerNorm: one row (H=1024) per 256-thread block, writes bf16
// ---------------------------------------------------------------------------
__global__ __launch_bounds__(256)
void layernorm_bf16_k(const float* __restrict__ x, const float* __restrict__ g,
                      const float* __restrict__ b, bf16* __restrict__ out) {
    __shared__ float rs[8], rq[8];
    const int row = blockIdx.x;
    const float* xr = x + (size_t)row * H_;
    float v[4], s = 0.f, sq = 0.f;
#pragma unroll
    for (int i = 0; i < 4; ++i) {
        v[i] = xr[threadIdx.x + i * 256];
        s  += v[i];
        sq += v[i] * v[i];
    }
#pragma unroll
    for (int o = 1; o < 32; o <<= 1) {
        s  += __shfl_xor(s,  o, 32);
        sq += __shfl_xor(sq, o, 32);
    }
    const int wid = threadIdx.x >> 5;
    if ((threadIdx.x & 31) == 0) { rs[wid] = s; rq[wid] = sq; }
    __syncthreads();
    float ts = 0.f, tq = 0.f;
#pragma unroll
    for (int i = 0; i < 8; ++i) { ts += rs[i]; tq += rq[i]; }
    const float mu  = ts * (1.f / H_);
    const float var = tq * (1.f / H_) - mu * mu;
    const float inv = rsqrtf(var + EPS_);
#pragma unroll
    for (int i = 0; i < 4; ++i) {
        int c = threadIdx.x + i * 256;
        out[(size_t)row * H_ + c] = (bf16)((v[i] - mu) * inv * g[c] + b[c]);
    }
}

// ---------------------------------------------------------------------------
// Generic bf16 WMMA GEMM  C[M,N] = op(A) @ op(B)  with fused epilogues.
//   TA: A stored [K,M] row-major;  TB: B stored [N,K] row-major.
// Tile 128x64x32, 256 threads = 8 waves (4 along M x 2 along N), wave = 32x32.
// Double-buffered LDS. A (non-T) staged via global_load_async_to_lds_b128 into
// a fragment-swizzled layout; B (non-T) staged via the Tensor Data Mover with
// D#-padded rows; transposed operands use VGPR staging + gather.
// MODE epilogues:
//   1: bf16 = acc + bias        2: f32+bf16 = acc + bias + auxF (residual)
//   3: bf16 = relu(acc + bias)  4: bf16 = (acc + bias - auxF) * scale
//   5: bf16 = acc * (auxB > 0)  6: f32 = acc
//   7: f32 = acc + bias + auxF
// ---------------------------------------------------------------------------
template <int MODE, bool TA, bool TB>
__global__ __launch_bounds__(256)
void gemm_bf16_wmma(const bf16* __restrict__ A, const bf16* __restrict__ Bm,
                    int M, int N, int K,
                    const float* __restrict__ bias,
                    const float* __restrict__ auxF,
                    const bf16*  __restrict__ auxB,
                    float* __restrict__ outF,
                    bf16*  __restrict__ outB,
                    float scale) {
    constexpr int BM = 128, BN = 64, BK = 32;
    constexpr int BS = BN + 8;                    // 144B row stride (16B multiple)
    constexpr int AGS = BK + 8;                   // gather layout row stride (TA)
    __shared__ __align__(32) bf16 AswS[TA ? 1 : 2 * 8 * 32 * 16];
    __shared__ bf16 AgS[TA ? 2 * BM * AGS : 1];
    __shared__ bf16 BsS[2 * BK * BS];

    const int tid  = threadIdx.x;
    const int lane = tid & 31;
    const int ln   = lane & 15;
    const int hh   = lane >> 4;
    const int wid  = tid >> 5;
    const int wm   = wid & 3;                     // 4 waves along M
    const int wn   = wid >> 2;                    // 2 waves along N
    const int m0   = blockIdx.y * BM;
    const int n0   = blockIdx.x * BN;

    v8f acc[2][2] = {};

    auto stage = [&](int t, int buf) {
        const int ks = t * BK;
        if constexpr (!TA) {
            // chunk kc -> lane half h = kc&1, elem block c = kc>>1 (fragment swizzle)
            for (int v = tid; v < (BM * BK) / 8; v += 256) {
                const int m = v >> 2, kc = v & 3;
                const bf16* gp = A + (size_t)(m0 + m) * K + ks + kc * 8;
                const int slot = (((buf * 8 + (m >> 4)) * 32) +
                                  ((m & 15) + 16 * (kc & 1))) * 16 + (kc >> 1) * 8;
#if USE_ASYNC
                async_b128(&AswS[slot], gp);
#else
                *(bf16x8*)&AswS[slot] = *(const bf16x8*)gp;
#endif
            }
        } else {
            for (int v = tid; v < (BM * BK) / 8; v += 256) {
                const int kk = v >> 4, m = (v & 15) * 8;
                bf16x8 tt = *(const bf16x8*)(A + (size_t)(ks + kk) * M + m0 + m);
#pragma unroll
                for (int j = 0; j < 8; ++j) AgS[(buf * BM + m + j) * AGS + kk] = tt[j];
            }
        }
        if constexpr (!TB) {
#if HAVE_TDM
            if (wid == 0)
                tdm_load_tile_b(Bm + (size_t)ks * N + n0,
                                (unsigned)(uintptr_t)&BsS[buf * BK * BS],
                                (unsigned)N, (unsigned)K);
#else
            for (int v = tid; v < (BK * BN) / 8; v += 256) {
                const int kk = v >> 3, n = (v & 7) * 8;
                const bf16* gp = Bm + (size_t)(ks + kk) * N + n0 + n;
#if USE_ASYNC
                async_b128(&BsS[(buf * BK + kk) * BS + n], gp);
#else
                *(bf16x8*)&BsS[(buf * BK + kk) * BS + n] = *(const bf16x8*)gp;
#endif
            }
#endif
        } else {
            for (int v = tid; v < (BK * BN) / 8; v += 256) {
                const int n = v >> 2, kk = (v & 3) * 8;
                bf16x8 tt = *(const bf16x8*)(Bm + (size_t)(n0 + n) * K + ks + kk);
#pragma unroll
                for (int j = 0; j < 8; ++j) BsS[(buf * BK + kk + j) * BS + n] = tt[j];
            }
        }
    };

    constexpr bool ANY_ASYNC = USE_ASYNC && ((!TA) || (!TB && !HAVE_TDM));
    auto fences = [&]() {
        if constexpr (ANY_ASYNC) async_wait0();
#if HAVE_TDM
        if constexpr (!TB) {
            if (wid == 0) __builtin_amdgcn_s_wait_tensorcnt(0);
        }
#endif
        __syncthreads();
    };

    const int T = K / BK;
    stage(0, 0);
    fences();

    for (int t = 0; t < T; ++t) {
        const int cur = t & 1;
        if (t + 1 < T) stage(t + 1, cur ^ 1);

        v16bf af[2], bfr[2];
        if constexpr (!TA) {
#pragma unroll
            for (int mi = 0; mi < 2; ++mi)
                af[mi] = *(const v16bf*)&AswS[(((cur * 8 + wm * 2 + mi) * 32) + lane) * 16];
        } else {
#pragma unroll
            for (int mi = 0; mi < 2; ++mi) {
                const int mr = wm * 32 + mi * 16 + ln;
#pragma unroll
                for (int e = 0; e < 16; ++e)
                    af[mi][e] = AgS[(cur * BM + mr) * AGS + akmap(e, hh)];
            }
        }
#pragma unroll
        for (int ni = 0; ni < 2; ++ni) {
            const int nc = wn * 32 + ni * 16 + ln;
#pragma unroll
            for (int e = 0; e < 16; ++e)
                bfr[ni][e] = BsS[(cur * BK + akmap(e, hh)) * BS + nc];
        }
#pragma unroll
        for (int mi = 0; mi < 2; ++mi)
#pragma unroll
            for (int ni = 0; ni < 2; ++ni)
                acc[mi][ni] = __builtin_amdgcn_wmma_f32_16x16x32_bf16(
                    false, af[mi], false, bfr[ni], (short)0, acc[mi][ni], false, false);

        fences();
    }

    // ---- epilogue ----
#pragma unroll
    for (int mi = 0; mi < 2; ++mi) {
#pragma unroll
        for (int ni = 0; ni < 2; ++ni) {
            const int col = n0 + wn * 32 + ni * 16 + ln;
#pragma unroll
            for (int r = 0; r < 8; ++r) {
                const int row = m0 + wm * 32 + mi * 16 + r + 8 * hh;
                const size_t idx = (size_t)row * N + col;
                const float x = acc[mi][ni][r];
                if (MODE == 1) {
                    outB[idx] = (bf16)(x + bias[col]);
                } else if (MODE == 2) {
                    float y = x + bias[col] + auxF[idx];
                    outF[idx] = y; outB[idx] = (bf16)y;
                } else if (MODE == 3) {
                    float y = x + bias[col];
                    outB[idx] = (bf16)(y > 0.f ? y : 0.f);
                } else if (MODE == 4) {
                    outB[idx] = (bf16)((x + bias[col] - auxF[idx]) * scale);
                } else if (MODE == 5) {
                    outB[idx] = (bf16)(((float)auxB[idx] > 0.f) ? x : 0.f);
                } else if (MODE == 6) {
                    outF[idx] = x;
                } else if (MODE == 7) {
                    outF[idx] = x + bias[col] + auxF[idx];
                } else {
                    outB[idx] = (bf16)x;
                }
            }
        }
    }
}

// ---------------------------------------------------------------------------
// Fused persistent-memory attention (per head, P=64, HD=64).
// ---------------------------------------------------------------------------
__global__ __launch_bounds__(64)
void attn_persistent_k(const bf16* __restrict__ qb, const bf16* __restrict__ pvb,
                       bf16* __restrict__ attnb) {
    __shared__ bf16 kv[P_][HD_ + 2];
    __shared__ bf16 qs[32][HD_ + 2];
    __shared__ bf16 pr[32][P_ + 2];

    const int tid  = threadIdx.x;
    const int lane = tid & 31;
    const int ln   = lane & 15;
    const int hh   = lane >> 4;
    const int w    = tid >> 5;
    const int head = blockIdx.y;
    const int bidx = blockIdx.z;
    const int row0 = blockIdx.x * 32;

    for (int i = tid; i < P_ * HD_; i += 64)
        kv[i / HD_][i % HD_] = pvb[(size_t)(i / HD_) * H_ + head * HD_ + (i % HD_)];
    for (int i = tid; i < 32 * HD_; i += 64)
        qs[i / HD_][i % HD_] =
            qb[((size_t)bidx * S_ + row0 + i / HD_) * H_ + head * HD_ + (i % HD_)];
    __syncthreads();

    v8f sacc[4] = {};
#pragma unroll
    for (int k0 = 0; k0 < HD_; k0 += 32) {
        v16bf a;
        const int mr = 16 * w + ln;
#pragma unroll
        for (int e = 0; e < 16; ++e) a[e] = qs[mr][k0 + akmap(e, hh)];
#pragma unroll
        for (int f = 0; f < 4; ++f) {
            v16bf bb;
            const int pc = f * 16 + ln;
#pragma unroll
            for (int e = 0; e < 16; ++e) bb[e] = kv[pc][k0 + akmap(e, hh)];   // KV^T
            sacc[f] = __builtin_amdgcn_wmma_f32_16x16x32_bf16(
                false, a, false, bb, (short)0, sacc[f], false, false);
        }
    }

#pragma unroll
    for (int r = 0; r < 8; ++r) {
        float mx = -1e30f;
#pragma unroll
        for (int f = 0; f < 4; ++f) mx = fmaxf(mx, sacc[f][r] * 0.125f);
#pragma unroll
        for (int o = 1; o < 16; o <<= 1) mx = fmaxf(mx, __shfl_xor(mx, o, 32));
        float ex[4], sum = 0.f;
#pragma unroll
        for (int f = 0; f < 4; ++f) { ex[f] = __expf(sacc[f][r] * 0.125f - mx); sum += ex[f]; }
#pragma unroll
        for (int o = 1; o < 16; o <<= 1) sum += __shfl_xor(sum, o, 32);
        const float inv = 1.f / sum;
        const int prow = 16 * w + r + 8 * hh;
#pragma unroll
        for (int f = 0; f < 4; ++f) pr[prow][f * 16 + ln] = (bf16)(ex[f] * inv);
    }
    __syncthreads();

    v8f aacc[4] = {};
#pragma unroll
    for (int k0 = 0; k0 < P_; k0 += 32) {
        v16bf a;
        const int mr = 16 * w + ln;
#pragma unroll
        for (int e = 0; e < 16; ++e) a[e] = pr[mr][k0 + akmap(e, hh)];
#pragma unroll
        for (int f = 0; f < 4; ++f) {
            v16bf bb;
            const int dc = f * 16 + ln;
#pragma unroll
            for (int e = 0; e < 16; ++e) bb[e] = kv[k0 + akmap(e, hh)][dc];
            aacc[f] = __builtin_amdgcn_wmma_f32_16x16x32_bf16(
                false, a, false, bb, (short)0, aacc[f], false, false);
        }
    }
#pragma unroll
    for (int f = 0; f < 4; ++f)
#pragma unroll
        for (int r = 0; r < 8; ++r) {
            const size_t row = (size_t)bidx * S_ + row0 + 16 * w + r + 8 * hh;
            attnb[row * H_ + head * HD_ + f * 16 + ln] = (bf16)aacc[f][r];
        }
}

// ---------------------------------------------------------------------------
// Column sums (bias grads)
// ---------------------------------------------------------------------------
__global__ void colsum_bf16_k(const bf16* __restrict__ X, float* __restrict__ out,
                              int rows, int cols) {
    int c = blockIdx.x * blockDim.x + threadIdx.x;
    if (c >= cols) return;
    float a = 0.f;
    for (int r = 0; r < rows; ++r) a += (float)X[(size_t)r * cols + c];
    out[c] = a;
}

// ---------------------------------------------------------------------------
// SGD-with-momentum + weight-decay; optional bf16 mirror of updated param.
// ---------------------------------------------------------------------------
__global__ void sgd_update_k(const float* __restrict__ p, const float* __restrict__ m,
                             const float* __restrict__ g, float* __restrict__ np,
                             bf16* __restrict__ npb, int n) {
    int i = blockIdx.x * blockDim.x + threadIdx.x;
    if (i >= n) return;
    float nm = MOM_ * m[i] - LR_ * g[i];
    float v  = (1.f - WD_) * p[i] + nm;
    np[i] = v;
    if (npb) npb[i] = (bf16)v;
}

// ---------------------------------------------------------------------------
// Orchestration
// ---------------------------------------------------------------------------
extern "C" void kernel_launch(void* const* d_in, const int* in_sizes, int n_in,
                              void* d_out, int out_size, void* d_ws, size_t ws_size,
                              hipStream_t stream) {
    (void)in_sizes; (void)n_in; (void)out_size; (void)ws_size;

    const float* hs  = (const float*)d_in[0];
    const float* km  = (const float*)d_in[1];
    const float* vm  = (const float*)d_in[2];
    const float* pv  = (const float*)d_in[3];
    const float* Wq  = (const float*)d_in[4];
    const float* bq  = (const float*)d_in[5];
    const float* Wo  = (const float*)d_in[6];
    const float* bo  = (const float*)d_in[7];
    const float* lng = (const float*)d_in[8];
    const float* lnb = (const float*)d_in[9];
    const float* W1  = (const float*)d_in[10];
    const float* b1  = (const float*)d_in[11];
    const float* W2  = (const float*)d_in[12];
    const float* b2  = (const float*)d_in[13];
    const float* mW1 = (const float*)d_in[14];
    const float* mb1 = (const float*)d_in[15];
    const float* mW2 = (const float*)d_in[16];
    const float* mb2 = (const float*)d_in[17];
    float* outp = (float*)d_out;

    size_t off = 0;
    char* base = (char*)d_ws;
    auto carve = [&](size_t bytes) -> void* {
        void* p = base + off;
        off += (bytes + 255) & ~(size_t)255;
        return p;
    };
    bf16*  Wqb   = (bf16*)carve((size_t)H_ * H_ * 2);
    bf16*  Wob   = (bf16*)carve((size_t)H_ * H_ * 2);
    bf16*  W1b   = (bf16*)carve((size_t)H_ * INTER_ * 2);
    bf16*  W2b   = (bf16*)carve((size_t)INTER_ * H_ * 2);
    bf16*  pvb   = (bf16*)carve((size_t)P_ * H_ * 2);
    bf16*  kb    = (bf16*)carve((size_t)NROWS * H_ * 2);
    bf16*  hnb   = (bf16*)carve((size_t)NROWS * H_ * 2);     // reused as out1b
    bf16*  qb    = (bf16*)carve((size_t)NROWS * H_ * 2);     // reused as db
    bf16*  attnb = (bf16*)carve((size_t)NROWS * H_ * 2);
    float* out1f = (float*)carve((size_t)NROWS * H_ * 4);
    bf16*  hb    = (bf16*)carve((size_t)NROWS * INTER_ * 2); // reused as hfb
    bf16*  dzb   = (bf16*)carve((size_t)NROWS * INTER_ * 2);
    float* gW1f  = (float*)carve((size_t)H_ * INTER_ * 4);
    float* gW2f  = (float*)carve((size_t)INTER_ * H_ * 4);
    float* gb1   = (float*)carve((size_t)INTER_ * 4);
    float* gb2   = (float*)carve((size_t)H_ * 4);
    float* nW1f  = (float*)carve((size_t)H_ * INTER_ * 4);
    bf16*  nW1b  = (bf16*)carve((size_t)H_ * INTER_ * 2);
    float* nW2f  = (float*)carve((size_t)INTER_ * H_ * 4);
    bf16*  nW2b  = (bf16*)carve((size_t)INTER_ * H_ * 2);
    float* nb1   = (float*)carve((size_t)INTER_ * 4);
    float* nb2   = (float*)carve((size_t)H_ * 4);

    bf16* out1b = hnb;
    bf16* db    = qb;
    bf16* hfb   = hb;

    auto cvt = [&](const float* src, bf16* dst, int n) {
        f32_to_bf16_k<<<(n + 255) / 256, 256, 0, stream>>>(src, dst, n);
    };
    cvt(Wq, Wqb, H_ * H_);
    cvt(Wo, Wob, H_ * H_);
    cvt(W1, W1b, H_ * INTER_);
    cvt(W2, W2b, INTER_ * H_);
    cvt(pv, pvb, P_ * H_);
    cvt(km, kb, NROWS * H_);

    layernorm_bf16_k<<<NROWS, 256, 0, stream>>>(hs, lng, lnb, hnb);

    gemm_bf16_wmma<1, false, false><<<dim3(H_ / 64, NROWS / 128), 256, 0, stream>>>(
        hnb, Wqb, NROWS, H_, H_, bq, nullptr, nullptr, nullptr, qb, 0.f);

    attn_persistent_k<<<dim3(S_ / 32, NH_, B_), 64, 0, stream>>>(qb, pvb, attnb);

    gemm_bf16_wmma<2, false, false><<<dim3(H_ / 64, NROWS / 128), 256, 0, stream>>>(
        attnb, Wob, NROWS, H_, H_, bo, hs, nullptr, out1f, out1b, 0.f);

    gemm_bf16_wmma<3, false, false><<<dim3(INTER_ / 64, NROWS / 128), 256, 0, stream>>>(
        kb, W1b, NROWS, INTER_, H_, b1, nullptr, nullptr, nullptr, hb, 0.f);

    const float dscale = 2.f / ((float)NROWS * (float)H_);
    gemm_bf16_wmma<4, false, false><<<dim3(H_ / 64, NROWS / 128), 256, 0, stream>>>(
        hb, W2b, NROWS, H_, INTER_, b2, vm, nullptr, nullptr, db, dscale);

    colsum_bf16_k<<<H_ / 256, 256, 0, stream>>>(db, gb2, NROWS, H_);

    gemm_bf16_wmma<5, false, true><<<dim3(INTER_ / 64, NROWS / 128), 256, 0, stream>>>(
        db, W2b, NROWS, INTER_, H_, nullptr, nullptr, hb, nullptr, dzb, 0.f);

    colsum_bf16_k<<<INTER_ / 256, 256, 0, stream>>>(dzb, gb1, NROWS, INTER_);

    gemm_bf16_wmma<6, true, false><<<dim3(H_ / 64, INTER_ / 128), 256, 0, stream>>>(
        hb, db, INTER_, H_, NROWS, nullptr, nullptr, nullptr, gW2f, nullptr, 0.f);

    gemm_bf16_wmma<6, true, false><<<dim3(INTER_ / 64, H_ / 128), 256, 0, stream>>>(
        kb, dzb, H_, INTER_, NROWS, nullptr, nullptr, nullptr, gW1f, nullptr, 0.f);

    sgd_update_k<<<(H_ * INTER_ + 255) / 256, 256, 0, stream>>>(W1, mW1, gW1f, nW1f, nW1b, H_ * INTER_);
    sgd_update_k<<<(INTER_ * H_ + 255) / 256, 256, 0, stream>>>(W2, mW2, gW2f, nW2f, nW2b, INTER_ * H_);
    sgd_update_k<<<(INTER_ + 255) / 256, 256, 0, stream>>>(b1, mb1, gb1, nb1, nullptr, INTER_);
    sgd_update_k<<<(H_ + 255) / 256, 256, 0, stream>>>(b2, mb2, gb2, nb2, nullptr, H_);

    gemm_bf16_wmma<3, false, false><<<dim3(INTER_ / 64, NROWS / 128), 256, 0, stream>>>(
        out1b, nW1b, NROWS, INTER_, H_, nb1, nullptr, nullptr, nullptr, hfb, 0.f);

    gemm_bf16_wmma<7, false, false><<<dim3(H_ / 64, NROWS / 128), 256, 0, stream>>>(
        hfb, nW2b, NROWS, H_, INTER_, nb2, out1f, nullptr, outp, nullptr, 0.f);
}
